// IDASR_48979807043777
// MI455X (gfx1250) — matrix operand order repaired and checked by reading, not compiled, not run on hardware
//
#include <hip/hip_runtime.h>
#include <hip/hip_bf16.h>
#include <math.h>

// ---------------------------------------------------------------------------
// IDASR on MI455X (gfx1250): all GEMM-shaped work (convs-as-implicit-GEMM,
// 1x1 conv, 12546->256 MLP) runs on v_wmma_f32_16x16x32_f16 (wave32 WMMA,
// fp16 A/B, f32 accumulate). Each wave computes a 16x64 output tile: one A
// fragment is reused across 4 B fragments, and the 4 independent accumulators
// let the scheduler interleave WMMAs (hides the f16 WMMA->WMMA RAW hazard,
// ISA 7.12.1). Irregular sampling/softmax is VALU+LDS+shfl (wave32).
// ---------------------------------------------------------------------------

typedef _Float16 h8_t  __attribute__((ext_vector_type(8)));
typedef _Float16 v16h  __attribute__((ext_vector_type(16)));
typedef float    v8f   __attribute__((ext_vector_type(8)));

#define HH 64
#define WW 64
#define NPIX 4096
#define RA 49
#define DIMK 12544   // RA*256 (rel_cell handled in epilogue)

__device__ __forceinline__ h8_t ldh8(const _Float16* p) { return *(const h8_t*)p; }

__device__ __forceinline__ v16h mk_frag(h8_t lo, h8_t hi) {
  v16h f;
#pragma unroll
  for (int i = 0; i < 8; ++i) { f[i] = lo[i]; f[i + 8] = hi[i]; }
  return f;
}

// A-fragment (16x32 f16): lane row = lane&15; K halves at k0+8g and k0+16+8g
__device__ __forceinline__ v16h frag_A(const _Float16* row, int k0, int g) {
  return mk_frag(ldh8(row + k0 + 8 * g), ldh8(row + k0 + 16 + 8 * g));
}
// B-fragment (32x16 f16): lane col = lane&15; K halves at k0+8g and k0+16+8g
__device__ __forceinline__ v16h frag_B(const _Float16* col, int k0, int g) {
  return mk_frag(ldh8(col + k0 + 8 * g), ldh8(col + k0 + 16 + 8 * g));
}

// ---------------------------------------------------------------------------
// conv1: 3->64, 3x3 pad1, ReLU. Writes HWC fp16 directly (WMMA B operand).
// grid = 4096 pixels, block = 64 channels.
// ---------------------------------------------------------------------------
__global__ void conv_enc_kernel(const float* __restrict__ inp,
                                const float* __restrict__ W, const float* __restrict__ b,
                                _Float16* __restrict__ out_hwc) {
  int px = blockIdx.x, c = threadIdx.x;
  int y = px >> 6, x = px & 63;
  float acc = b[c];
#pragma unroll
  for (int ci = 0; ci < 3; ++ci)
#pragma unroll
    for (int ky = 0; ky < 3; ++ky)
#pragma unroll
      for (int kx = 0; kx < 3; ++kx) {
        int yy = y + ky - 1, xx = x + kx - 1;
        if (yy >= 0 && yy < HH && xx >= 0 && xx < WW)
          acc += W[((c * 3 + ci) * 3 + ky) * 3 + kx] * inp[ci * NPIX + yy * WW + xx];
      }
  out_hwc[(size_t)px * 64 + c] = (_Float16)fmaxf(acc, 0.0f);
}

// ---------------------------------------------------------------------------
// Weight repack: W[Co][Cin][3][3] f32 -> Wp[Co][kidx*Cin+ci] f16
// ---------------------------------------------------------------------------
__global__ void pack_w9_kernel(const float* __restrict__ W, _Float16* __restrict__ Wp,
                               int Co, int Cin) {
  int idx = blockIdx.x * blockDim.x + threadIdx.x;
  int total = Co * Cin * 9;
  if (idx >= total) return;
  int co = idx / (9 * Cin);
  int r = idx - co * 9 * Cin;
  int kidx = r / Cin, ci = r - kidx * Cin;
  Wp[idx] = (_Float16)W[((size_t)co * Cin + ci) * 9 + kidx];
}

// CHW f32 -> HWC f16
__global__ void pack_hwc_kernel(const float* __restrict__ src, _Float16* __restrict__ dst, int C) {
  int idx = blockIdx.x * blockDim.x + threadIdx.x;
  if (idx >= NPIX * C) return;
  int px = idx / C, c = idx - px * C;
  dst[idx] = (_Float16)src[(size_t)c * NPIX + px];
}

// Wi1 (256 x 12546) f32 -> fp16 [256][12544] (last 2 cols live in the epilogue)
__global__ void pack_wi1_kernel(const float* __restrict__ W, _Float16* __restrict__ Wp) {
  int idx = blockIdx.x * blockDim.x + threadIdx.x;
  if (idx >= 256 * DIMK) return;
  int m = idx / DIMK, k = idx - m * DIMK;
  Wp[idx] = (_Float16)W[(size_t)m * 12546 + k];
}

// W_off2 (98 x 256) f32 -> fp16 [112][256], rows 98..111 zero
__global__ void pack_woff2_kernel(const float* __restrict__ W, _Float16* __restrict__ Wp) {
  int idx = blockIdx.x * blockDim.x + threadIdx.x;
  if (idx >= 112 * 256) return;
  int m = idx / 256, k = idx - m * 256;
  Wp[idx] = (m < 98) ? (_Float16)W[m * 256 + k] : (_Float16)0.0f;
}

// ---------------------------------------------------------------------------
// 3x3 conv as implicit GEMM via WMMA. One wave per 16(co) x 64(px) strip:
// the strip is one image row, A (weight) fragment reused across 4 pixel
// tiles, 4 independent accumulators interleave WMMAs.
// grid = (Co/16, 64), block = 32.
// ---------------------------------------------------------------------------
__global__ void conv3x3_wmma_kernel(const _Float16* __restrict__ Bhwc,
                                    const _Float16* __restrict__ Wp,
                                    const float* __restrict__ bias,
                                    float* __restrict__ out, int Cin) {
  int m0 = blockIdx.x * 16;
  int p0 = blockIdx.y * 64;          // one full image row
  int y = p0 >> 6;
  int lane = threadIdx.x, g = lane >> 4, ln = lane & 15;
  const _Float16* arow = Wp + (size_t)(m0 + ln) * (9 * Cin);
  v8f acc0 = {}, acc1 = {}, acc2 = {}, acc3 = {};
#pragma unroll
  for (int kidx = 0; kidx < 9; ++kidx) {
    int dy = kidx / 3 - 1, dx = kidx % 3 - 1;
    int yy = y + dy;
    bool yok = (yy >= 0 && yy < HH);
    const _Float16* brow[4];
    bool valid[4];
#pragma unroll
    for (int i = 0; i < 4; ++i) {
      int xx = 16 * i + ln + dx;
      valid[i] = yok && (xx >= 0) && (xx < WW);
      brow[i] = valid[i] ? (Bhwc + (size_t)(yy * WW + xx) * Cin) : nullptr;
    }
    for (int c0 = 0; c0 < Cin; c0 += 32) {
      v16h a = frag_A(arow, kidx * Cin + c0, g);
      v16h z = {};
      v16h b0 = valid[0] ? frag_B(brow[0], c0, g) : z;
      v16h b1 = valid[1] ? frag_B(brow[1], c0, g) : z;
      v16h b2 = valid[2] ? frag_B(brow[2], c0, g) : z;
      v16h b3 = valid[3] ? frag_B(brow[3], c0, g) : z;
      acc0 = __builtin_amdgcn_wmma_f32_16x16x32_f16(false, a, false, b0, (short)0, acc0, false, false);
      acc1 = __builtin_amdgcn_wmma_f32_16x16x32_f16(false, a, false, b1, (short)0, acc1, false, false);
      acc2 = __builtin_amdgcn_wmma_f32_16x16x32_f16(false, a, false, b2, (short)0, acc2, false, false);
      acc3 = __builtin_amdgcn_wmma_f32_16x16x32_f16(false, a, false, b3, (short)0, acc3, false, false);
    }
  }
  v8f accs[4] = {acc0, acc1, acc2, acc3};
#pragma unroll
  for (int i = 0; i < 4; ++i)
#pragma unroll
    for (int r = 0; r < 8; ++r) {
      int m = m0 + r + 8 * g;
      float v = accs[i][r] + (bias ? bias[m] : 0.0f);
      out[(size_t)m * NPIX + p0 + 16 * i + ln] = v;
    }
}

// ---------------------------------------------------------------------------
// Generic GEMM: out[M][N] = A[M][K](f16) * X[N][K]^T(f16) + bias (+ cell cols)
// One wave per 16(M) x 64(N) tile; A fragment reused across 4 X fragments.
// Used for: 1x1 conv (M=112 padded->98) and the 12544-K MLP (fused ReLU +
// rel_cell term from Wi1 columns 12544/12545). grid = (M/16, N/64), block=32.
// ---------------------------------------------------------------------------
__global__ void gemm_wmma_kernel(const _Float16* __restrict__ A,
                                 const _Float16* __restrict__ X,
                                 const float* __restrict__ bias,
                                 const float* __restrict__ wcA,  // col 12544 base (stride wstride)
                                 const float* __restrict__ wcB,  // col 12545 base
                                 const float* __restrict__ cell, // [N][2]
                                 float* __restrict__ out,
                                 int N, int K, int Mreal, int wstride, int relu) {
  int m0 = blockIdx.x * 16, n0 = blockIdx.y * 64;
  int lane = threadIdx.x, g = lane >> 4, ln = lane & 15;
  const _Float16* arow = A + (size_t)(m0 + ln) * K;
  const _Float16* xr0 = X + (size_t)(n0 + 0  + ln) * K;
  const _Float16* xr1 = X + (size_t)(n0 + 16 + ln) * K;
  const _Float16* xr2 = X + (size_t)(n0 + 32 + ln) * K;
  const _Float16* xr3 = X + (size_t)(n0 + 48 + ln) * K;
  v8f acc0 = {}, acc1 = {}, acc2 = {}, acc3 = {};
  for (int k0 = 0; k0 < K; k0 += 32) {
    v16h a = frag_A(arow, k0, g);
    v16h b0 = frag_B(xr0, k0, g);
    v16h b1 = frag_B(xr1, k0, g);
    v16h b2 = frag_B(xr2, k0, g);
    v16h b3 = frag_B(xr3, k0, g);
    acc0 = __builtin_amdgcn_wmma_f32_16x16x32_f16(false, a, false, b0, (short)0, acc0, false, false);
    acc1 = __builtin_amdgcn_wmma_f32_16x16x32_f16(false, a, false, b1, (short)0, acc1, false, false);
    acc2 = __builtin_amdgcn_wmma_f32_16x16x32_f16(false, a, false, b2, (short)0, acc2, false, false);
    acc3 = __builtin_amdgcn_wmma_f32_16x16x32_f16(false, a, false, b3, (short)0, acc3, false, false);
  }
  v8f accs[4] = {acc0, acc1, acc2, acc3};
#pragma unroll
  for (int i = 0; i < 4; ++i) {
    int n = n0 + 16 * i + ln;
    float rc0 = 0.0f, rc1 = 0.0f;
    if (cell) {
      rc0 = cell[(size_t)n * 2 + 0] * (float)HH;
      rc1 = cell[(size_t)n * 2 + 1] * (float)WW;
    }
#pragma unroll
    for (int r = 0; r < 8; ++r) {
      int m = m0 + r + 8 * g;
      if (m < Mreal) {
        float v = accs[i][r];
        if (bias) v += bias[m];
        if (wcA)  v += wcA[(size_t)m * wstride] * rc0 + wcB[(size_t)m * wstride] * rc1;
        if (relu) v = fmaxf(v, 0.0f);
        out[(size_t)m * N + n] = v;
      }
    }
  }
}

// ---------------------------------------------------------------------------
// Grouped 5x5 conv (groups=8), direct. grid = 4096 px, block = 256 co.
// ---------------------------------------------------------------------------
__global__ void gconv5x5_kernel(const float* __restrict__ feat,
                                const float* __restrict__ W, const float* __restrict__ b,
                                float* __restrict__ out) {
  int px = blockIdx.x, co = threadIdx.x;
  int y = px >> 6, x = px & 63;
  int gi = co >> 5;
  float acc = b[co];
  for (int ci = 0; ci < 32; ++ci) {
    const float* fch = feat + (size_t)(gi * 32 + ci) * NPIX;
    const float* wch = W + ((size_t)co * 32 + ci) * 25;
#pragma unroll
    for (int ky = 0; ky < 5; ++ky) {
      int yy = y + ky - 2;
      if (yy < 0 || yy >= HH) continue;
#pragma unroll
      for (int kx = 0; kx < 5; ++kx) {
        int xx = x + kx - 2;
        if (xx < 0 || xx >= WW) continue;
        acc += wch[ky * 5 + kx] * fch[yy * WW + xx];
      }
    }
  }
  out[(size_t)co * NPIX + px] = acc;
}

// ---------------------------------------------------------------------------
// LayerNorm over 256 channels + exact GELU, writes HWC fp16 (GEMM B operand).
// grid = 4096 px, block = 256.
// ---------------------------------------------------------------------------
__global__ void ln_gelu_kernel(const float* __restrict__ o,
                               const float* __restrict__ gam, const float* __restrict__ bet,
                               _Float16* __restrict__ act_hwc) {
  __shared__ float red[256];
  int px = blockIdx.x, c = threadIdx.x;
  float v = o[(size_t)c * NPIX + px];
  red[c] = v; __syncthreads();
  for (int s = 128; s > 0; s >>= 1) { if (c < s) red[c] += red[c + s]; __syncthreads(); }
  float mu = red[0] * (1.0f / 256.0f); __syncthreads();
  float d = v - mu;
  red[c] = d * d; __syncthreads();
  for (int s = 128; s > 0; s >>= 1) { if (c < s) red[c] += red[c + s]; __syncthreads(); }
  float var = red[0] * (1.0f / 256.0f);
  float xn = d * rsqrtf(var + 1e-5f) * gam[c] + bet[c];
  float ge = 0.5f * xn * (1.0f + erff(xn * 0.70710678118654752f));
  act_hwc[(size_t)px * 256 + c] = (_Float16)ge;
}

// ---------------------------------------------------------------------------
// Per-query deformable sampling + attention; writes attn-weighted v as the
// fp16 MLP operand Xfeat[q][12544]. One 256-thread block per query;
// wave = head (wave32), lane = dim; q.k reduced with shfl_xor.
// ---------------------------------------------------------------------------
__global__ void sample_attn_kernel(const float* __restrict__ coord,
                                   const float* __restrict__ offs,  // [98][4096] f32
                                   const float* __restrict__ fq, const float* __restrict__ fk,
                                   const float* __restrict__ fv,
                                   const float* __restrict__ Wc1, const float* __restrict__ bc1,
                                   const float* __restrict__ Wc2, const float* __restrict__ bc2,
                                   _Float16* __restrict__ Xfeat) {
  __shared__ float qv[256], hid[256], rel[98], wc[RA];
  __shared__ float logits[RA * 8], attnw[RA * 8], mj[RA];
  __shared__ int pj[RA];
  int q = blockIdx.x, t = threadIdx.x;
  int h = t >> 5, lane = t & 31;
  float sy = coord[q * 2 + 0], sx = coord[q * 2 + 1];
  float fx = ((sx + 1.0f) * (float)WW - 1.0f) * 0.5f;
  float fy = ((sy + 1.0f) * (float)HH - 1.0f) * 0.5f;
  int ix = (int)rintf(fx), iy = (int)rintf(fy);
  float bmask = (ix >= 0 && ix < WW && iy >= 0 && iy < HH) ? 1.0f : 0.0f;
  int ixc = min(max(ix, 0), WW - 1), iyc = min(max(iy, 0), HH - 1);
  int bp = iyc * WW + ixc;
  float base_y = (-1.0f + 1.0f / HH + (2.0f / HH) * iyc) * bmask;
  float base_x = (-1.0f + 1.0f / WW + (2.0f / WW) * ixc) * bmask;

  if (t < RA) {
    int a = t / 7, b = t % 7;
    float offy = offs[(size_t)(2 * t) * NPIX + bp] * bmask;
    float offx = offs[(size_t)(2 * t + 1) * NPIX + bp] * bmask;
    float sy_j = base_y + (a - 3) * (2.0f / HH) + tanhf(offy) * (2.0f / (HH - 1.0f));
    float sx_j = base_x + (b - 3) * (2.0f / WW) + tanhf(offx) * (2.0f / (WW - 1.0f));
    rel[2 * t] = (sy - sy_j) * (float)HH;
    rel[2 * t + 1] = (sx - sx_j) * (float)WW;
    float jfx = ((sx_j + 1.0f) * WW - 1.0f) * 0.5f;
    float jfy = ((sy_j + 1.0f) * HH - 1.0f) * 0.5f;
    int jx = (int)rintf(jfx), jy = (int)rintf(jfy);
    mj[t] = (jx >= 0 && jx < WW && jy >= 0 && jy < HH) ? 1.0f : 0.0f;
    pj[t] = min(max(jy, 0), HH - 1) * WW + min(max(jx, 0), WW - 1);
  }
  // q: bilinear, zero padding; channel = t
  {
    float x0f = floorf(fx), y0f = floorf(fy);
    float wx = fx - x0f, wy = fy - y0f;
    int x0 = (int)x0f, y0 = (int)y0f;
    float acc = 0.0f;
#pragma unroll
    for (int dy = 0; dy < 2; ++dy)
#pragma unroll
      for (int dx = 0; dx < 2; ++dx) {
        int xi = x0 + dx, yi = y0 + dy;
        if (xi >= 0 && xi < WW && yi >= 0 && yi < HH) {
          float w = (dx ? wx : 1.0f - wx) * (dy ? wy : 1.0f - wy);
          acc += w * fq[(size_t)t * NPIX + yi * WW + xi];
        }
      }
    qv[t] = acc;
  }
  __syncthreads();

  for (int j = 0; j < RA; ++j) {
    float kv = mj[j] * fk[(size_t)t * NPIX + pj[j]];
    float p = qv[t] * kv;
#pragma unroll
    for (int off = 16; off > 0; off >>= 1) p += __shfl_xor(p, off, 32);
    if (lane == 0) logits[j * 8 + h] = p * 0.17677669529663687f;  // 1/sqrt(32)
  }
  __syncthreads();

  // wcoord MLP: 98 -> 256 -> 49
  {
    float s = bc1[t];
    for (int i = 0; i < 98; ++i) s += Wc1[(size_t)t * 98 + i] * rel[i];
    hid[t] = fmaxf(s, 0.0f);
  }
  __syncthreads();
  if (t < RA) {
    float s = bc2[t];
    for (int i = 0; i < 256; ++i) s += Wc2[(size_t)t * 256 + i] * hid[i];
    wc[t] = s;
  }
  __syncthreads();

  if (t < 8) {
    float mx = -1e30f;
    for (int j = 0; j < RA; ++j) {
      float l = logits[j * 8 + t] + wc[j];
      attnw[j * 8 + t] = l;
      mx = fmaxf(mx, l);
    }
    float s = 0.0f;
    for (int j = 0; j < RA; ++j) {
      float e = __expf(attnw[j * 8 + t] - mx);
      attnw[j * 8 + t] = e; s += e;
    }
    float inv = 1.0f / s;
    for (int j = 0; j < RA; ++j) attnw[j * 8 + t] *= inv;
  }
  __syncthreads();

  _Float16* xrow = Xfeat + (size_t)q * DIMK;
  for (int j = 0; j < RA; ++j) {
    float vv = mj[j] * fv[(size_t)t * NPIX + pj[j]];
    xrow[j * 256 + t] = (_Float16)(vv * attnw[j * 8 + h]);
  }
}

// ---------------------------------------------------------------------------
// Final: pred = Wi2(3x256) * hidden + bi2, plus border-bilinear skip of inp.
// ---------------------------------------------------------------------------
__global__ void final_kernel(const float* __restrict__ hidden,
                             const float* __restrict__ Wi2, const float* __restrict__ bi2,
                             const float* __restrict__ inp, const float* __restrict__ coord,
                             float* __restrict__ out) {
  int q = blockIdx.x * blockDim.x + threadIdx.x;
  if (q >= NPIX) return;
  float sy = coord[q * 2 + 0], sx = coord[q * 2 + 1];
  float gx = fminf(fmaxf(((sx + 1.0f) * WW - 1.0f) * 0.5f, 0.0f), (float)(WW - 1));
  float gy = fminf(fmaxf(((sy + 1.0f) * HH - 1.0f) * 0.5f, 0.0f), (float)(HH - 1));
  float x0f = floorf(gx), y0f = floorf(gy);
  float wx = gx - x0f, wy = gy - y0f;
  int x0 = (int)x0f, y0 = (int)y0f;
  float wts[4]; int pix[4];
#pragma unroll
  for (int dy = 0; dy < 2; ++dy)
#pragma unroll
    for (int dx = 0; dx < 2; ++dx) {
      int xi = min(max(x0 + dx, 0), WW - 1);
      int yi = min(max(y0 + dy, 0), HH - 1);
      wts[dy * 2 + dx] = (dx ? wx : 1.0f - wx) * (dy ? wy : 1.0f - wy);
      pix[dy * 2 + dx] = yi * WW + xi;
    }
#pragma unroll
  for (int c = 0; c < 3; ++c) {
    float p = bi2[c];
    for (int k = 0; k < 256; ++k) p += Wi2[c * 256 + k] * hidden[(size_t)k * NPIX + q];
    float sk = 0.0f;
#pragma unroll
    for (int i = 0; i < 4; ++i) sk += wts[i] * inp[(size_t)c * NPIX + pix[i]];
    out[(size_t)q * 3 + c] = p + sk;
  }
}

// ---------------------------------------------------------------------------
extern "C" void kernel_launch(void* const* d_in, const int* in_sizes, int n_in,
                              void* d_out, int out_size, void* d_ws, size_t ws_size,
                              hipStream_t stream) {
  const float* inp    = (const float*)d_in[0];
  const float* coord  = (const float*)d_in[1];
  const float* cell   = (const float*)d_in[2];
  const float* W_enc  = (const float*)d_in[3];
  const float* b_enc  = (const float*)d_in[4];
  const float* W_ch   = (const float*)d_in[5];
  const float* b_ch   = (const float*)d_in[6];
  const float* W_q    = (const float*)d_in[7];
  const float* b_q    = (const float*)d_in[8];
  const float* W_k    = (const float*)d_in[9];
  const float* b_k    = (const float*)d_in[10];
  const float* W_v    = (const float*)d_in[11];
  const float* b_v    = (const float*)d_in[12];
  const float* W_off1 = (const float*)d_in[13];
  const float* b_off1 = (const float*)d_in[14];
  const float* ln_g   = (const float*)d_in[15];
  const float* ln_b   = (const float*)d_in[16];
  const float* W_off2 = (const float*)d_in[17];
  const float* Wc1    = (const float*)d_in[18];
  const float* bc1    = (const float*)d_in[19];
  const float* Wc2    = (const float*)d_in[20];
  const float* bc2    = (const float*)d_in[21];
  const float* Wi1    = (const float*)d_in[22];
  const float* bi1    = (const float*)d_in[23];
  const float* Wi2    = (const float*)d_in[24];
  const float* bi2    = (const float*)d_in[25];
  float* out = (float*)d_out;

  // workspace carve-up (256B aligned); total ~= 138 MiB
  char* base = (char*)d_ws;
  size_t off = 0;
  auto alloc = [&](size_t bytes) {
    char* p = base + off;
    off += (bytes + 255) & ~(size_t)255;
    return p;
  };
  _Float16* off0hwc = (_Float16*)alloc((size_t)NPIX * 64 * 2);
  _Float16* Wp_ch   = (_Float16*)alloc((size_t)256 * 9 * 64 * 2);
  _Float16* Wp_q    = (_Float16*)alloc((size_t)256 * 9 * 256 * 2);
  _Float16* Wp_k    = (_Float16*)alloc((size_t)256 * 9 * 256 * 2);
  _Float16* Wp_v    = (_Float16*)alloc((size_t)256 * 9 * 256 * 2);
  _Float16* Wp2     = (_Float16*)alloc((size_t)112 * 256 * 2);
  _Float16* Wi1h    = (_Float16*)alloc((size_t)256 * DIMK * 2);
  float*    feat    = (float*)alloc((size_t)256 * NPIX * 4);
  _Float16* featHWC = (_Float16*)alloc((size_t)NPIX * 256 * 2);
  float*    fq      = (float*)alloc((size_t)256 * NPIX * 4);
  float*    fk      = (float*)alloc((size_t)256 * NPIX * 4);
  float*    fv      = (float*)alloc((size_t)256 * NPIX * 4);
  float*    obuf    = (float*)alloc((size_t)256 * NPIX * 4);
  _Float16* actHWC  = (_Float16*)alloc((size_t)NPIX * 256 * 2);
  float*    offsets = (float*)alloc((size_t)98 * NPIX * 4);
  _Float16* Xfeat   = (_Float16*)alloc((size_t)NPIX * DIMK * 2);
  float*    hidden  = (float*)alloc((size_t)256 * NPIX * 4);
  (void)ws_size; (void)n_in; (void)in_sizes; (void)out_size;

  // weight / activation repacks (fp32 -> WMMA-friendly fp16 layouts)
  pack_w9_kernel<<<(256 * 64 * 9 + 255) / 256, 256, 0, stream>>>(W_ch, Wp_ch, 256, 64);
  pack_w9_kernel<<<(256 * 256 * 9 + 255) / 256, 256, 0, stream>>>(W_q, Wp_q, 256, 256);
  pack_w9_kernel<<<(256 * 256 * 9 + 255) / 256, 256, 0, stream>>>(W_k, Wp_k, 256, 256);
  pack_w9_kernel<<<(256 * 256 * 9 + 255) / 256, 256, 0, stream>>>(W_v, Wp_v, 256, 256);
  pack_woff2_kernel<<<(112 * 256 + 255) / 256, 256, 0, stream>>>(W_off2, Wp2);
  pack_wi1_kernel<<<(256 * DIMK + 255) / 256, 256, 0, stream>>>(Wi1, Wi1h);

  // encoder
  conv_enc_kernel<<<NPIX, 64, 0, stream>>>(inp, W_enc, b_enc, off0hwc);
  conv3x3_wmma_kernel<<<dim3(256 / 16, NPIX / 64), 32, 0, stream>>>(off0hwc, Wp_ch, b_ch, feat, 64);
  pack_hwc_kernel<<<(NPIX * 256 + 255) / 256, 256, 0, stream>>>(feat, featHWC, 256);

  // q/k/v convs (WMMA implicit GEMM, 16x64 tiles)
  conv3x3_wmma_kernel<<<dim3(16, 64), 32, 0, stream>>>(featHWC, Wp_q, b_q, fq, 256);
  conv3x3_wmma_kernel<<<dim3(16, 64), 32, 0, stream>>>(featHWC, Wp_k, b_k, fk, 256);
  conv3x3_wmma_kernel<<<dim3(16, 64), 32, 0, stream>>>(featHWC, Wp_v, b_v, fv, 256);

  // offset branch: grouped conv -> LN+GELU -> 1x1 conv (WMMA GEMM)
  gconv5x5_kernel<<<NPIX, 256, 0, stream>>>(feat, W_off1, b_off1, obuf);
  ln_gelu_kernel<<<NPIX, 256, 0, stream>>>(obuf, ln_g, ln_b, actHWC);
  gemm_wmma_kernel<<<dim3(112 / 16, NPIX / 64), 32, 0, stream>>>(
      Wp2, actHWC, nullptr, nullptr, nullptr, nullptr, offsets, NPIX, 256, 98, 0, 0);

  // deformable sampling + attention -> fp16 MLP operand
  sample_attn_kernel<<<NPIX, 256, 0, stream>>>(coord, offsets, fq, fk, fv,
                                               Wc1, bc1, Wc2, bc2, Xfeat);

  // big MLP 12546->256 (dominant 26 GFLOP): WMMA GEMM + fused rel_cell + ReLU
  gemm_wmma_kernel<<<dim3(256 / 16, NPIX / 64), 32, 0, stream>>>(
      Wi1h, Xfeat, bi1, Wi1 + DIMK, Wi1 + DIMK + 1, cell, hidden, NPIX, DIMK, 256, 12546, 1);

  // 256->3 projection + border-bilinear skip
  final_kernel<<<NPIX / 256, 256, 0, stream>>>(hidden, Wi2, bi2, inp, coord, out);
}
